// FlowMambaCell_54820962566285
// MI455X (gfx1250) — matrix operand order repaired and checked by reading, not compiled, not run on hardware
//
#include <hip/hip_runtime.h>
#include <hip/hip_bf16.h>
#include <cmath>

#define D_MODEL 96
#define D_STATE 16
#define IMG     32
#define HW      1024
#define NUM_V   5
#define GROUPS  4
#define BATCH   4
#define KTOT    864      // 96 channels * 9 taps
#define NOUT    128      // 96 (delta) + 16 (B) + 16 (C)
#define KSTEPS  27       // 864 / 32

typedef __attribute__((ext_vector_type(16))) _Float16 v16h;
typedef __attribute__((ext_vector_type(8)))  _Float16 v8h;
typedef __attribute__((ext_vector_type(8)))  float    v8f;

// ---------------------------------------------------------------------------
// 1) GroupNorm: per (b, group) reduction, writes u_norm as f16 [b][h][w][c]
// ---------------------------------------------------------------------------
__global__ void fm_gn_kernel(const float* __restrict__ u,
                             const float* __restrict__ gamma,
                             const float* __restrict__ beta,
                             _Float16* __restrict__ uN) {
  __shared__ float s1[256];
  __shared__ float s2[256];
  const int b = blockIdx.x / GROUPS;
  const int g = blockIdx.x % GROUPS;
  const int CG = D_MODEL / GROUPS;   // 24
  const int NE = CG * HW;            // 24576
  const float* base = u + ((size_t)b * D_MODEL + (size_t)g * CG) * HW;

  float acc = 0.f, acc2 = 0.f;
  for (int i = threadIdx.x; i < NE; i += 256) {
    float v = base[i];
    acc += v; acc2 += v * v;
  }
  s1[threadIdx.x] = acc; s2[threadIdx.x] = acc2;
  __syncthreads();
  for (int st = 128; st > 0; st >>= 1) {
    if ((int)threadIdx.x < st) {
      s1[threadIdx.x] += s1[threadIdx.x + st];
      s2[threadIdx.x] += s2[threadIdx.x + st];
    }
    __syncthreads();
  }
  const float inv = 1.0f / (float)NE;
  const float mu = s1[0] * inv;
  const float var = s2[0] * inv - mu * mu;
  const float rstd = rsqrtf(var + 1e-5f);

  for (int i = threadIdx.x; i < NE; i += 256) {
    int c_l = i >> 10, pix = i & 1023;
    int c = g * CG + c_l;
    float xn = (base[i] - mu) * rstd * gamma[c] + beta[c];
    uN[((size_t)(b * HW + pix)) * D_MODEL + c] = (_Float16)xn;
  }
}

// ---------------------------------------------------------------------------
// 2) im2col with circular wrap: Acol[b][pixel][t*96 + c] (f16)
// ---------------------------------------------------------------------------
__global__ void fm_im2col_kernel(const _Float16* __restrict__ uN,
                                 _Float16* __restrict__ Acol) {
  int idx = blockIdx.x * blockDim.x + threadIdx.x;   // BATCH*HW*9
  if (idx >= BATCH * HW * 9) return;
  int t = idx % 9;
  int p = (idx / 9) % HW;
  int b = idx / (9 * HW);
  int h = p >> 5, w = p & 31;
  int hh = (h + (t / 3) - 1) & 31;
  int ww = (w + (t % 3) - 1) & 31;
  const uint4* src = (const uint4*)(uN + ((size_t)(b * HW + (hh << 5) + ww)) * D_MODEL);
  uint4* dst = (uint4*)(Acol + ((size_t)(b * HW + p)) * KTOT + (size_t)t * D_MODEL);
#pragma unroll
  for (int i = 0; i < 12; ++i) dst[i] = src[i];     // 96 halves = 192 B
}

// ---------------------------------------------------------------------------
// 3) Pack combined [K=864][N=128] weights into per-lane WMMA B-fragment order:
//    Wfrag[((nt*27+ks)*32+lane)*16 + i]  where K = ks*32 + (lane>>4)*16 + i,
//    N = nt*16 + (lane&15).  K = t*96 + c.
// ---------------------------------------------------------------------------
__global__ void fm_wpack_kernel(const float* __restrict__ wD,
                                const float* __restrict__ wB,
                                const float* __restrict__ wC,
                                _Float16* __restrict__ Wfrag) {
  int idx = blockIdx.x * blockDim.x + threadIdx.x;   // 8*27*32
  if (idx >= (NOUT / 16) * KSTEPS * 32) return;
  int lane = idx & 31;
  int ks = (idx >> 5) % KSTEPS;
  int nt = idx / (32 * KSTEPS);
  int n = nt * 16 + (lane & 15);
  int kb = ks * 32 + (lane >> 4) * 16;
  _Float16* dst = Wfrag + (size_t)idx * 16;
#pragma unroll
  for (int i = 0; i < 16; ++i) {
    int k = kb + i;
    int t = k / D_MODEL, c = k % D_MODEL;
    float v;
    if (n < D_MODEL)                 v = wD[((size_t)n * D_MODEL + c) * 9 + t];
    else if (n < D_MODEL + D_STATE)  v = wB[((size_t)(n - D_MODEL) * D_MODEL + c) * 9 + t];
    else                             v = wC[((size_t)(n - D_MODEL - D_STATE) * D_MODEL + c) * 9 + t];
    dst[i] = (_Float16)v;
  }
}

// ---------------------------------------------------------------------------
// 4) Implicit-GEMM conv via WMMA: M=16 pixels x N=16 outs per wave, K=864.
//    Block = 256 threads = 8 waves covering all 128 output columns of 1 M-tile.
//    Epilogue: softplus/clip -> delta ; raw -> B_val, C_val.
// ---------------------------------------------------------------------------
__global__ void fm_conv_wmma_kernel(const _Float16* __restrict__ Acol,
                                    const _Float16* __restrict__ Wfrag,
                                    const float* __restrict__ b_delta,
                                    const float* __restrict__ dt_p,
                                    float* __restrict__ deltaW,
                                    float* __restrict__ BvW,
                                    float* __restrict__ CvW) {
  const int b = blockIdx.x >> 6;       // 4 batches
  const int ptile = blockIdx.x & 63;   // 64 tiles of 16 pixels
  const int lane = threadIdx.x & 31;
  const int nt = threadIdx.x >> 5;     // N-tile 0..7
  const int m = lane & 15;             // A row (pixel in tile)
  const int khalf = lane >> 4;         // K half select

  // A: contiguous K order matching B pack, each lane needs one run of 16 halves
  const _Float16* arow = Acol + ((size_t)(b * HW + ptile * 16 + m)) * KTOT + khalf * 16;
  // B: pre-packed fragment order -> one contiguous 16-half load per lane
  const _Float16* brow = Wfrag + ((size_t)nt * KSTEPS) * 32 * 16 + (size_t)lane * 16;

  v8f acc = {};
  for (int ks = 0; ks < KSTEPS; ++ks) {
    v8h a0 = *(const v8h*)(arow + ks * 32);        // K = kb .. kb+7
    v8h a1 = *(const v8h*)(arow + ks * 32 + 8);    // K = kb+8 .. kb+15
    v16h a;
#pragma unroll
    for (int i = 0; i < 8; ++i) { a[i] = a0[i]; a[8 + i] = a1[i]; }
    v16h bf = *(const v16h*)(brow + (size_t)ks * 32 * 16);
    acc = __builtin_amdgcn_wmma_f32_16x16x32_f16(
        /*neg_a=*/false, a, /*neg_b=*/false, bf,
        /*c_mod=*/(short)0, acc, /*reuse_a=*/false, /*reuse_b=*/false);
  }

  const float dt = dt_p[0];
  const int n = nt * 16 + (lane & 15);
#pragma unroll
  for (int r = 0; r < 8; ++r) {
    int p = ptile * 16 + khalf * 8 + r;            // C/D: M = (lane>>4)*8 + r
    float v = acc[r];
    if (n < D_MODEL) {
      float x = v + b_delta[n] + dt;
      float sp = (x > 15.f) ? x : log1pf(__expf(x));
      sp = fminf(fmaxf(sp, 1e-4f), 5.0f);
      deltaW[((size_t)(b * D_MODEL + n)) * HW + p] = sp;
    } else if (n < D_MODEL + D_STATE) {
      BvW[((size_t)(b * D_STATE + (n - D_MODEL))) * HW + p] = v;
    } else {
      CvW[((size_t)(b * D_STATE + (n - D_MODEL - D_STATE))) * HW + p] = v;
    }
  }
}

// ---------------------------------------------------------------------------
// 5) Fused streaming state update: the HBM-bound kernel.
//    Thread = (b,d,h,w). Loop n: compute A_bar/B_bar once, reuse across 5
//    roll directions; s_prev read once, s_new written once, y in registers.
// ---------------------------------------------------------------------------
__global__ void fm_update_kernel(const float* __restrict__ u_t,
                                 const float* __restrict__ s_prev,
                                 const float* __restrict__ log_A,
                                 const float* __restrict__ Dp,
                                 const float* __restrict__ deltaW,
                                 const float* __restrict__ BvW,
                                 const float* __restrict__ CvW,
                                 float* __restrict__ y_out,
                                 float* __restrict__ s_new) {
  const int bid = blockIdx.x;                 // BATCH * D_MODEL * 4
  const int hg = bid & 3;
  const int d = (bid >> 2) % D_MODEL;
  const int b = bid / (4 * D_MODEL);
  const int h = hg * 8 + (threadIdx.x >> 5);
  const int w = threadIdx.x & 31;
  const int p = (h << 5) | w;

  const size_t up = ((size_t)(b * D_MODEL + d)) * HW + p;
  const float ut = u_t[up];
  const float dl = deltaW[up];

  float y[NUM_V] = {0.f, 0.f, 0.f, 0.f, 0.f};
  const int vh[NUM_V] = {0, 0, 0, 1, -1};
  const int vw[NUM_V] = {0, 1, -1, 0, 0};

  const size_t sV = (size_t)D_MODEL * D_STATE * HW;               // per-direction stride
  const size_t sBD = (((size_t)b * NUM_V) * D_MODEL + d) * (size_t)D_STATE * HW;

  for (int n = 0; n < D_STATE; ++n) {
    const float Abar = __expf(-__expf(log_A[d * D_STATE + n]) * dl);
    const float bu = dl * BvW[((size_t)(b * D_STATE + n)) * HW + p] * ut;
    const float cv = CvW[((size_t)(b * D_STATE + n)) * HW + p];
    const size_t sN = sBD + (size_t)n * HW;
    if (n + 1 < D_STATE)
      __builtin_prefetch((const void*)(s_prev + sN + HW + p), 0, 3);
#pragma unroll
    for (int v = 0; v < NUM_V; ++v) {
      const int hs = (h - vh[v]) & 31;
      const int ws = (w - vw[v]) & 31;
      const float sp = s_prev[sN + (size_t)v * sV + (size_t)((hs << 5) | ws)];
      const float sn = Abar * sp + bu;
      s_new[sN + (size_t)v * sV + p] = sn;
      y[v] += sn * cv;
    }
  }

  const float du = ut * Dp[d];
#pragma unroll
  for (int v = 0; v < NUM_V; ++v)
    y_out[(((size_t)(b * NUM_V + v)) * D_MODEL + d) * HW + p] = y[v] + du;
}

// ---------------------------------------------------------------------------
extern "C" void kernel_launch(void* const* d_in, const int* in_sizes, int n_in,
                              void* d_out, int out_size, void* d_ws, size_t ws_size,
                              hipStream_t stream) {
  (void)in_sizes; (void)n_in; (void)out_size; (void)ws_size;
  const float* u_t     = (const float*)d_in[0];
  const float* s_prev  = (const float*)d_in[1];
  const float* gamma   = (const float*)d_in[2];
  const float* beta    = (const float*)d_in[3];
  const float* w_delta = (const float*)d_in[4];
  const float* b_delta = (const float*)d_in[5];
  const float* w_B     = (const float*)d_in[6];
  const float* w_C     = (const float*)d_in[7];
  const float* log_A   = (const float*)d_in[8];
  const float* Dp      = (const float*)d_in[9];
  const float* dtp     = (const float*)d_in[10];

  char* ws = (char*)d_ws;
  size_t off = 0;
  auto alloc = [&](size_t bytes) -> char* {
    char* pp = ws + off;
    off = (off + bytes + 255) & ~(size_t)255;
    return pp;
  };
  _Float16* uN    = (_Float16*)alloc((size_t)BATCH * HW * D_MODEL * 2);
  _Float16* Acol  = (_Float16*)alloc((size_t)BATCH * HW * KTOT * 2);
  _Float16* Wfrag = (_Float16*)alloc((size_t)(NOUT / 16) * KSTEPS * 32 * 16 * 2);
  float* deltaW   = (float*)alloc((size_t)BATCH * D_MODEL * HW * 4);
  float* BvW      = (float*)alloc((size_t)BATCH * D_STATE * HW * 4);
  float* CvW      = (float*)alloc((size_t)BATCH * D_STATE * HW * 4);

  float* y_out = (float*)d_out;
  float* s_new = y_out + (size_t)BATCH * NUM_V * D_MODEL * HW;

  fm_gn_kernel<<<BATCH * GROUPS, 256, 0, stream>>>(u_t, gamma, beta, uN);
  fm_im2col_kernel<<<(BATCH * HW * 9 + 255) / 256, 256, 0, stream>>>(uN, Acol);
  fm_wpack_kernel<<<((NOUT / 16) * KSTEPS * 32 + 255) / 256, 256, 0, stream>>>(w_delta, w_B, w_C, Wfrag);
  fm_conv_wmma_kernel<<<BATCH * 64, 256, 0, stream>>>(Acol, Wfrag, b_delta, dtp, deltaW, BvW, CvW);
  fm_update_kernel<<<BATCH * D_MODEL * 4, 256, 0, stream>>>(u_t, s_prev, log_A, Dp,
                                                            deltaW, BvW, CvW, y_out, s_new);
}
